// DLRMv2_18176301597439
// MI455X (gfx1250) — matrix operand order repaired and checked by reading, not compiled, not run on hardware
//
#include <hip/hip_runtime.h>
#include <hip/hip_bf16.h>

// ---------------------------------------------------------------------------
// DLRM-v2 forward for MI455X (gfx1250, wave32).
//   * all GEMMs: v_wmma_f32_16x16x32_bf16, f32 accumulate
//   * bf16 activations + pre-padded bf16 weights -> guard-free, K%32==0 GEMMs
//   * tile staging: global_load_async_to_lds_b128 (ASYNCcnt), double-buffered
//     so chunk k+1 DMAs while chunk k runs on the WMMA pipe
// ---------------------------------------------------------------------------

#define GEMM_ASYNC_COPY 1   // set 0 to fall back to load->ds_store staging

typedef __attribute__((ext_vector_type(16))) __bf16 v16bf;
typedef __attribute__((ext_vector_type(8)))  float  v8f;

#define NUM_TABLES 26
#define VOCAB      10000
#define EMB_D      128

// round-to-nearest-even f32 -> bf16 (bit pattern)
__device__ __forceinline__ unsigned short f2bf(float x) {
    unsigned u = __builtin_bit_cast(unsigned, x);
    u += 0x7fffu + ((u >> 16) & 1u);
    return (unsigned short)(u >> 16);
}
__device__ __forceinline__ unsigned f2bf2(float lo, float hi) {
    return (unsigned)f2bf(lo) | ((unsigned)f2bf(hi) << 16);
}
__device__ __forceinline__ float bf2f(unsigned short s) {
    unsigned u = ((unsigned)s) << 16;
    return __builtin_bit_cast(float, u);
}

// --- WMMA fragment loaders (layouts per CDNA5 ISA 7.12.2, wave32) ----------
// A: 16x32 bf16. lanes 0-15: K 0-7 (v0-3), 16-23 (v4-7); lanes16-31: K 8-15, 24-31
// (contiguous 16B runs -> compiler merges into ds_load_b128 x2)
__device__ __forceinline__ v16bf load_frag_a(const unsigned short* base,
                                             int stride, int mBase, int kOff,
                                             int lane) {
    const int m    = mBase + (lane & 15);
    const int half = lane >> 4;
    const unsigned short* row = base + m * stride + kOff;
    union { unsigned u[8]; v16bf v; } f;
#pragma unroll
    for (int i = 0; i < 4; ++i) {
        f.u[i]     = *(const unsigned*)(row + (half * 8 + 2 * i));
        f.u[4 + i] = *(const unsigned*)(row + (16 + half * 8 + 2 * i));
    }
    return f.v;
}

// B: 32x16 (K x N) sourced from W's N-major rows: lanes 0-15 K 0-7, 16-23;
// lanes 16-31 K 8-15, 24-31 (same K order as A).
__device__ __forceinline__ v16bf load_frag_b(const unsigned short* base,
                                             int stride, int nBase, int kOff,
                                             int lane) {
    const int n    = nBase + (lane & 15);
    const int half = lane >> 4;
    const unsigned short* row = base + n * stride + kOff;
    union { unsigned u[8]; v16bf v; } f;
#pragma unroll
    for (int i = 0; i < 4; ++i) {
        f.u[i]     = *(const unsigned*)(row + (half * 8 + 2 * i));
        f.u[4 + i] = *(const unsigned*)(row + (16 + half * 8 + 2 * i));
    }
    return f.v;
}

// 16B-per-lane tile copy: global bf16 -> LDS (async DMA or through-VGPR)
__device__ __forceinline__ void copy16_g2l(unsigned short* ldst,
                                           const unsigned short* gsrc) {
#if GEMM_ASYNC_COPY
    const unsigned loff = (unsigned)(__SIZE_TYPE__)(const void*)ldst;
    asm volatile("global_load_async_to_lds_b128 %0, %1, off"
                 :: "v"(loff), "v"(gsrc) : "memory");
#else
    *(uint4*)ldst = *(const uint4*)gsrc;
#endif
}
__device__ __forceinline__ void wait_async_le4() {
#if GEMM_ASYNC_COPY
    asm volatile("s_wait_asynccnt 0x4" ::: "memory");
#endif
}
__device__ __forceinline__ void wait_async_0() {
#if GEMM_ASYNC_COPY
    asm volatile("s_wait_asynccnt 0x0" ::: "memory");
#endif
}

// ---------------------------------------------------------------------------
// Y(bf16, MxN) = act(A(bf16, M x lda) @ W(bf16, N x ldw)^T + bias(f32)).
// K % 32 == 0, lda/ldw % 8 == 0 (16B rows).  128 thr = 4 waves, 64x64 tile,
// 32x32 per wave (2x2 WMMA acc), K chunk 32, double-buffered async staging.
// ---------------------------------------------------------------------------
#define GT_S 40   // LDS row stride (ushorts): 80B rows -> 16B aligned, no conflicts

__global__ __launch_bounds__(128) void gemm_bf16_wmma(
    const unsigned short* __restrict__ A, int lda,
    const unsigned short* __restrict__ W, int ldw,
    const float* __restrict__ bias,
    unsigned short* __restrict__ Y,
    int M, int N, int K, int doRelu)
{
    __shared__ __align__(16) unsigned short lds_a[2][64 * GT_S];
    __shared__ __align__(16) unsigned short lds_b[2][64 * GT_S];

    const int tid  = threadIdx.x;
    const int lane = tid & 31;
    const int wave = tid >> 5;
    const int m0   = blockIdx.y * 64;
    const int n0   = blockIdx.x * 64;
    const int wm   = (wave >> 1) * 32;
    const int wn   = (wave & 1)  * 32;

    // per-thread staging coordinates: 64 rows x 4 x 16B chunks, 2 per thread
    const int row0 = tid >> 2;                 // rows 0..31   (p=0)
    const int row1 = (tid + 128) >> 2;         // rows 32..63  (p=1)
    const int c8   = (tid & 3) * 8;            // 8 bf16 = 16B

    auto stage = [&](int buf, int k0) {
        copy16_g2l(lds_a[buf] + row0 * GT_S + c8,
                   A + (size_t)(m0 + row0) * lda + k0 + c8);
        copy16_g2l(lds_b[buf] + row0 * GT_S + c8,
                   W + (size_t)(n0 + row0) * ldw + k0 + c8);
        copy16_g2l(lds_a[buf] + row1 * GT_S + c8,
                   A + (size_t)(m0 + row1) * lda + k0 + c8);
        copy16_g2l(lds_b[buf] + row1 * GT_S + c8,
                   W + (size_t)(n0 + row1) * ldw + k0 + c8);
    };

    v8f acc[2][2] = {};
    const int nk = K >> 5;

    stage(0, 0);                               // prefetch chunk 0
    for (int kc = 0; kc < nk; ++kc) {
        const int cur = kc & 1;
        if (kc + 1 < nk) {                     // uniform branch
            stage(cur ^ 1, (kc + 1) << 5);     // prefetch next chunk
            wait_async_le4();                  // in-order -> current chunk done
        } else {
            wait_async_0();
        }
        __syncthreads();                       // everyone's copies visible

#pragma unroll
        for (int i = 0; i < 2; ++i) {
            const v16bf af = load_frag_a(lds_a[cur], GT_S, wm + i * 16, 0, lane);
#pragma unroll
            for (int j = 0; j < 2; ++j) {
                const v16bf bf = load_frag_b(lds_b[cur], GT_S, wn + j * 16, 0, lane);
                acc[i][j] = __builtin_amdgcn_wmma_f32_16x16x32_bf16(
                    false, af, false, bf, (short)0, acc[i][j], false, false);
            }
        }
        __syncthreads();                       // done reading buf before reuse
    }

    // C/D layout: element (m = e + 8*half, n = lane&15)
    const int cn   = lane & 15;
    const int half = lane >> 4;
#pragma unroll
    for (int i = 0; i < 2; ++i) {
#pragma unroll
        for (int j = 0; j < 2; ++j) {
            const int colg = n0 + wn + j * 16 + cn;
            const float bv = bias[colg];
#pragma unroll
            for (int e = 0; e < 8; ++e) {
                const int rowg = m0 + wm + i * 16 + e + half * 8;
                float v = acc[i][j][e] + bv;
                if (doRelu) v = fmaxf(v, 0.0f);
                Y[(size_t)rowg * N + colg] = f2bf(v);
            }
        }
    }
}

// ---------------------------------------------------------------------------
// Pad+convert f32 (N x K) -> bf16 (N x Kp), zero-filled for k >= K.
// ---------------------------------------------------------------------------
__global__ __launch_bounds__(256) void convert_pad_bf16(
    const float* __restrict__ src, unsigned short* __restrict__ dst,
    int N, int K, int Kp)
{
    const int i = blockIdx.x * 256 + threadIdx.x;
    if (i >= N * Kp) return;
    const int n = i / Kp;
    const int k = i - n * Kp;
    dst[i] = (k < K) ? f2bf(src[(size_t)n * K + k]) : (unsigned short)0;
}

// ---------------------------------------------------------------------------
// Embedding gather + pairwise interaction.  4 waves/block, 1 sample/wave.
// E (32x128 bf16, rows 27..31 zero) in LDS; G = E E^T via WMMA tiles
// (0,0),(0,1),(1,1).  xtop[b] = [dense(128) | triu(G)(351) | 0] bf16, ld 480.
// ---------------------------------------------------------------------------
#define IE_S 132   // LDS row stride (ushorts): 264B rows, 8B aligned

__global__ __launch_bounds__(128) void embed_interact_wmma(
    const unsigned short* __restrict__ dense_out,  // M x 128 bf16
    const int*   __restrict__ sparse_x,            // M x 26
    const float* __restrict__ tables,              // 26 x 10000 x 128 f32
    unsigned short* __restrict__ xtop,             // M x 480 bf16
    int M)
{
    __shared__ __align__(16) unsigned short lds_e[4 * 32 * IE_S];

    const int lane = threadIdx.x & 31;
    const int wave = threadIdx.x >> 5;
    const int b    = blockIdx.x * 4 + wave;
    unsigned short* E = lds_e + wave * 32 * IE_S;

    const int c = lane * 4;   // 4 columns per lane
    // row 0: dense features (already bf16); also passed through to xtop
    const uint2 dv = *(const uint2*)(dense_out + (size_t)b * EMB_D + c);
    *(unsigned*)(E + c)     = dv.x;
    *(unsigned*)(E + c + 2) = dv.y;
    {
        unsigned short* xd = xtop + (size_t)b * 480;
        *(unsigned*)(xd + c)     = dv.x;
        *(unsigned*)(xd + c + 2) = dv.y;
        if (lane == 0) xd[479] = 0;            // zero the K-pad column
    }
    // rows 1..26: gathered embedding rows (f32 -> bf16, packed u32 stores)
#pragma unroll 1
    for (int r = 1; r <= NUM_TABLES; ++r) {
        const int t   = r - 1;
        const int idx = sparse_x[(size_t)b * NUM_TABLES + t];
        const float4 v =
            *(const float4*)(tables + ((size_t)t * VOCAB + idx) * EMB_D + c);
        unsigned short* dst = E + r * IE_S + c;
        *(unsigned*)(dst)     = f2bf2(v.x, v.y);
        *(unsigned*)(dst + 2) = f2bf2(v.z, v.w);
    }
    // rows 27..31: zero pad
#pragma unroll
    for (int r = NUM_TABLES + 1; r < 32; ++r) {
        unsigned short* dst = E + r * IE_S + c;
        *(unsigned*)(dst)     = 0u;
        *(unsigned*)(dst + 2) = 0u;
    }
    __syncthreads();

    // Gram tiles covering triu(k=1) of the 27x27 matrix
    const int tmL[3] = {0, 0, 1};
    const int tnL[3] = {0, 1, 1};
    const int cn   = lane & 15;
    const int half = lane >> 4;

#pragma unroll
    for (int t = 0; t < 3; ++t) {
        const int tm = tmL[t], tn = tnL[t];
        v8f acc = {};
#pragma unroll
        for (int kc = 0; kc < 4; ++kc) {
            const v16bf af = load_frag_a(E, IE_S, tm * 16, kc * 32, lane);
            const v16bf bf = load_frag_b(E, IE_S, tn * 16, kc * 32, lane);
            acc = __builtin_amdgcn_wmma_f32_16x16x32_bf16(
                false, af, false, bf, (short)0, acc, false, false);
        }
#pragma unroll
        for (int e = 0; e < 8; ++e) {
            const int r  = tm * 16 + e + half * 8;
            const int cc = tn * 16 + cn;
            if (cc > r && cc < NUM_TABLES + 1) {
                const int flat = EMB_D + r * NUM_TABLES - (r * (r - 1)) / 2
                               + (cc - r - 1);
                xtop[(size_t)b * 480 + flat] = f2bf(acc[e]);
            }
        }
    }
}

// ---------------------------------------------------------------------------
// Final layer: out[b] = sigmoid(dot(x[b](bf16,256), tw2(f32)) + tb2).
// ---------------------------------------------------------------------------
__global__ __launch_bounds__(256) void top_final_sigmoid(
    const unsigned short* __restrict__ X,  // M x 256 bf16
    const float* __restrict__ w,           // 256 f32
    const float* __restrict__ bias,        // 1
    float* __restrict__ out, int M)
{
    __shared__ float lw[256];
    lw[threadIdx.x] = w[threadIdx.x];
    __syncthreads();

    const int b = blockIdx.x * 256 + threadIdx.x;
    if (b >= M) return;
    const uint4* xr = (const uint4*)(X + (size_t)b * 256);  // 8 bf16 / uint4
    float s = 0.0f;
#pragma unroll
    for (int i = 0; i < 32; ++i) {
        const uint4 v = xr[i];
        const float* wv = lw + i * 8;
        s = fmaf(bf2f((unsigned short)(v.x & 0xffff)), wv[0], s);
        s = fmaf(bf2f((unsigned short)(v.x >> 16)),    wv[1], s);
        s = fmaf(bf2f((unsigned short)(v.y & 0xffff)), wv[2], s);
        s = fmaf(bf2f((unsigned short)(v.y >> 16)),    wv[3], s);
        s = fmaf(bf2f((unsigned short)(v.z & 0xffff)), wv[4], s);
        s = fmaf(bf2f((unsigned short)(v.z >> 16)),    wv[5], s);
        s = fmaf(bf2f((unsigned short)(v.w & 0xffff)), wv[6], s);
        s = fmaf(bf2f((unsigned short)(v.w >> 16)),    wv[7], s);
    }
    s += bias[0];
    out[b] = 1.0f / (1.0f + expf(-s));
}

// ---------------------------------------------------------------------------
extern "C" void kernel_launch(void* const* d_in, const int* in_sizes, int n_in,
                              void* d_out, int out_size, void* d_ws, size_t ws_size,
                              hipStream_t stream) {
    const float* dense_x  = (const float*)d_in[0];
    const int*   sparse_x = (const int*)  d_in[1];
    // d_in[2] = sparse_offsets (arange(B) -> segment-sum is identity; unused)
    const float* tables   = (const float*)d_in[3];
    const float* bw0 = (const float*)d_in[4];  const float* bb0 = (const float*)d_in[5];
    const float* bw1 = (const float*)d_in[6];  const float* bb1 = (const float*)d_in[7];
    const float* bw2 = (const float*)d_in[8];  const float* bb2 = (const float*)d_in[9];
    const float* tw0 = (const float*)d_in[10]; const float* tb0 = (const float*)d_in[11];
    const float* tw1 = (const float*)d_in[12]; const float* tb1 = (const float*)d_in[13];
    const float* tw2 = (const float*)d_in[14]; const float* tb2 = (const float*)d_in[15];

    const int M = in_sizes[0] / 13;   // batch (32768)

    // -------- bf16 workspace layout (all 16B aligned) --------
    unsigned short* ws   = (unsigned short*)d_ws;
    unsigned short* dxb  = ws;                         // M x 32   (dense, padded)
    unsigned short* h0   = dxb  + (size_t)M * 32;      // M x 512
    unsigned short* h1   = h0   + (size_t)M * 512;     // M x 256
    unsigned short* dout = h1   + (size_t)M * 256;     // M x 128
    unsigned short* xtop = dout + (size_t)M * 128;     // M x 480
    unsigned short* bw0b = xtop + (size_t)M * 480;     // 512 x 32
    unsigned short* bw1b = bw0b + 512 * 32;            // 256 x 512
    unsigned short* bw2b = bw1b + 256 * 512;           // 128 x 256
    unsigned short* tw0b = bw2b + 128 * 256;           // 512 x 480
    unsigned short* tw1b = tw0b + 512 * 480;           // 256 x 512
    unsigned short* t0   = h0;                         // reuse dead buffers
    unsigned short* t1   = h1;

    // -------- pad/convert inputs + weights to bf16 --------
    auto cvt = [&](const float* s, unsigned short* d, int N, int K, int Kp) {
        const int tot = N * Kp;
        convert_pad_bf16<<<(tot + 255) / 256, 256, 0, stream>>>(s, d, N, K, Kp);
    };
    cvt(dense_x, dxb, M, 13, 32);
    cvt(bw0, bw0b, 512, 13, 32);
    cvt(bw1, bw1b, 256, 512, 512);
    cvt(bw2, bw2b, 128, 256, 256);
    cvt(tw0, tw0b, 512, 479, 480);
    cvt(tw1, tw1b, 256, 512, 512);

    const dim3 blk(128);

    // Bottom MLP: 13(->32) -> 512 -> 256 -> 128, ReLU
    gemm_bf16_wmma<<<dim3(512 / 64, M / 64), blk, 0, stream>>>(
        dxb, 32, bw0b, 32, bb0, h0, M, 512, 32, 1);
    gemm_bf16_wmma<<<dim3(256 / 64, M / 64), blk, 0, stream>>>(
        h0, 512, bw1b, 512, bb1, h1, M, 256, 512, 1);
    gemm_bf16_wmma<<<dim3(128 / 64, M / 64), blk, 0, stream>>>(
        h1, 256, bw2b, 256, bb2, dout, M, 128, 256, 1);

    // Gather + interaction -> xtop (M x 479 used, ld 480, col 479 zeroed)
    embed_interact_wmma<<<M / 4, 128, 0, stream>>>(
        dout, sparse_x, tables, xtop, M);

    // Top MLP: 479(->480) -> 512 -> 256 -> 1 (+ sigmoid)
    gemm_bf16_wmma<<<dim3(512 / 64, M / 64), blk, 0, stream>>>(
        xtop, 480, tw0b, 480, tb0, t0, M, 512, 480, 1);
    gemm_bf16_wmma<<<dim3(256 / 64, M / 64), blk, 0, stream>>>(
        t0, 512, tw1b, 512, tb1, t1, M, 256, 512, 1);
    top_final_sigmoid<<<M / 256, 256, 0, stream>>>(
        t1, tw2, tb2, (float*)d_out, M);

    (void)n_in; (void)out_size; (void)ws_size;
}
